// Block_14972255994199
// MI455X (gfx1250) — matrix operand order repaired and checked by reading, not compiled, not run on hardware
//
#include <hip/hip_runtime.h>
#include <hip/hip_bf16.h>
#include <math.h>

// ---------------------------------------------------------------------------
// CDNA5 (gfx1250) transformer block: LN -> QKV GEMM -> flash attention ->
// proj GEMM (+res) -> LN -> fc1 GEMM -> dwconv3x3+GELU -> fc2 GEMM (+res).
// Matmuls: v_wmma_f32_16x16x32_bf16 (wave32).
// Global->LDS staging: global_load_async_to_lds_b128 + s_wait_asynccnt,
// double-buffered (one barrier per K-step, fetch overlapped with WMMA).
// Fragment loads forced to ds_load_b128 via explicit uint4 accesses.
// ---------------------------------------------------------------------------

typedef __attribute__((ext_vector_type(16))) __bf16 v16bf;
typedef __attribute__((ext_vector_type(8)))  float  v8f;

__device__ __forceinline__ unsigned short f2bf(float f) {
  unsigned int u = __float_as_uint(f);
  u += 0x7fffu + ((u >> 16) & 1u);          // round-to-nearest-even
  return (unsigned short)(u >> 16);
}
__device__ __forceinline__ float bf2f(unsigned short h) {
  return __uint_as_float(((unsigned int)h) << 16);
}

// Async DMA of 16 bytes global -> LDS (ASYNCcnt path, bypasses VGPRs).
// lds_off: byte address within LDS (low 32 bits of a generic LDS pointer).
__device__ __forceinline__ void async_copy_b128(unsigned int lds_off,
                                                const void* g) {
  asm volatile("global_load_async_to_lds_b128 %0, %1, off"
               :: "v"(lds_off), "v"((unsigned long long)(size_t)g)
               : "memory");
}
__device__ __forceinline__ void wait_async0() {
  asm volatile("s_wait_asynccnt 0x0" ::: "memory");
}
__device__ __forceinline__ unsigned int lds_off_of(const void* p) {
  return (unsigned int)(size_t)p;
}

union FragBF { v16bf v; uint4 q[2]; };

// A-matrix fragment, 16x32 bf16 (ISA 7.12.2): lanes 0-15 M=0..15 / K-low half,
// lanes 16-31 same M / K-high half; VGPR v holds packed pair at
// k0 = (v<4 ? 2v + 8*half : 16 + 2(v-4) + 8*half).
// Per lane that is bytes [16*half,+16) and [32+16*half,+16) of the row ->
// two ds_load_b128 (explicit uint4 so the compiler must coalesce).
__device__ __forceinline__ v16bf load_a_frag(const unsigned short* base, int ld) {
  const int lane = threadIdx.x & 31;
  const int half = lane >> 4, m = lane & 15;
  const uint4* p = (const uint4*)(base + m * ld);
  FragBF r;
  r.q[0] = p[half];
  r.q[1] = p[2 + half];
  return r.v;
}

// B-matrix fragment, 32x16 bf16: lane n = lane&15, same K order as A:
// VGPR v holds pair at k0 = (v<4 ? 2v + 8*half : 16 + 2(v-4) + 8*half).
// `base` points at LDS laid out [n][k] (k contiguous), ld = k-stride.
// Per lane: bytes [16*half,+16) and [32+16*half,+16) of the row -> two ds_load_b128.
__device__ __forceinline__ v16bf load_b_frag(const unsigned short* base, int ld) {
  const int lane = threadIdx.x & 31;
  const int half = lane >> 4, n = lane & 15;
  const uint4* p = (const uint4*)(base + n * ld);
  FragBF r;
  r.q[0] = p[half];
  r.q[1] = p[2 + half];
  return r.v;
}

__device__ __forceinline__ v8f wmma_bf16(v16bf a, v16bf b, v8f c) {
  return __builtin_amdgcn_wmma_f32_16x16x32_bf16(false, a, false, b,
                                                 (short)0, c, false, false);
}

// ---------------------------------------------------------------------------
// fp32 -> bf16 conversion (weights)
// ---------------------------------------------------------------------------
__global__ __launch_bounds__(256) void cvt_bf16_kernel(
    const float* __restrict__ src, unsigned short* __restrict__ dst, int n) {
  int i = blockIdx.x * 256 + threadIdx.x;
  if (i < n) dst[i] = f2bf(src[i]);
}

// ---------------------------------------------------------------------------
// LayerNorm over C=512, one row per block (256 threads = 8 wave32), bf16 out
// ---------------------------------------------------------------------------
__global__ __launch_bounds__(256) void ln_bf16_kernel(
    const float* __restrict__ x, const float* __restrict__ g,
    const float* __restrict__ b, unsigned short* __restrict__ out) {
  const int row = blockIdx.x;
  const int t = threadIdx.x;
  const float* xr = x + (size_t)row * 512;
  float v0 = xr[t], v1 = xr[t + 256];
  float s = v0 + v1;
  float q = v0 * v0 + v1 * v1;
#pragma unroll
  for (int o = 16; o; o >>= 1) {
    s += __shfl_xor(s, o, 32);
    q += __shfl_xor(q, o, 32);
  }
  __shared__ float ps[8], pq[8];
  if ((t & 31) == 0) { ps[t >> 5] = s; pq[t >> 5] = q; }
  __syncthreads();
  float S = 0.f, Q = 0.f;
#pragma unroll
  for (int i = 0; i < 8; ++i) { S += ps[i]; Q += pq[i]; }
  const float mu  = S * (1.f / 512.f);
  const float var = Q * (1.f / 512.f) - mu * mu;
  const float inv = rsqrtf(var + 1e-5f);
  unsigned short* orow = out + (size_t)row * 512;
  orow[t]       = f2bf((v0 - mu) * inv * g[t] + b[t]);
  orow[t + 256] = f2bf((v1 - mu) * inv * g[t + 256] + b[t + 256]);
}

// ---------------------------------------------------------------------------
// Tiled bf16 GEMM:  Y[M,N] = A[M,K] * W[N,K]^T (+bias[N]) (+res[M,N])
// 128x128 block tile, K-step 32, double-buffered LDS filled by
// global_load_async_to_lds_b128; 8 waves each own 4x2 16x16 fp32 accums.
// ---------------------------------------------------------------------------
__global__ __launch_bounds__(256) void gemm_bf16(
    const unsigned short* __restrict__ A,
    const unsigned short* __restrict__ W,
    const float* __restrict__ bias,
    const float* __restrict__ res,
    float* __restrict__ outf,
    unsigned short* __restrict__ outh,
    int M, int N, int K) {
  __shared__ unsigned short As[2][128 * 32];   // 2 x 8KB
  __shared__ unsigned short Bs[2][128 * 32];   // 2 x 8KB

  const int t = threadIdx.x;
  const int mTile = blockIdx.y * 128;
  const int nTile = blockIdx.x * 128;
  const int w  = t >> 5;
  const int wm = w >> 2;       // 0..1  (64-row strip)
  const int wn = w & 3;        // 0..3  (32-col strip)
  const int lane = t & 31;
  const int half = lane >> 4;
  const int nn   = lane & 15;

  const int lr = t >> 1;            // staging row 0..127
  const int ls = (t & 1) * 16;      // staging 16-elem segment

  const v8f vz = {0.f, 0.f, 0.f, 0.f, 0.f, 0.f, 0.f, 0.f};
  v8f acc[4][2];
#pragma unroll
  for (int i = 0; i < 4; ++i)
#pragma unroll
    for (int j = 0; j < 2; ++j) acc[i][j] = vz;

  const size_t arow = (size_t)(mTile + lr) * K;
  const size_t brow = (size_t)(nTile + lr) * K;

  unsigned int aoff[2], boff[2];
#pragma unroll
  for (int b2 = 0; b2 < 2; ++b2) {
    aoff[b2] = lds_off_of(&As[b2][lr * 32 + ls]);
    boff[b2] = lds_off_of(&Bs[b2][lr * 32 + ls]);
  }

  // prologue: stage K-tile 0 into buffer 0 (4 async b128 per thread)
  async_copy_b128(aoff[0],      A + arow + ls);
  async_copy_b128(aoff[0] + 16, A + arow + ls + 8);
  async_copy_b128(boff[0],      W + brow + ls);
  async_copy_b128(boff[0] + 16, W + brow + ls + 8);

  const int nIter = K >> 5;
  for (int it = 0; it < nIter; ++it) {
    const int cur = it & 1;
    wait_async0();          // this wave's async writes for buf[cur] landed
    __syncthreads();        // everyone's writes visible; prev reads retired
    if (it + 1 < nIter) {   // overlap next tile's DMA with this tile's WMMAs
      const int nxt = cur ^ 1;
      const int kk = (it + 1) << 5;
      async_copy_b128(aoff[nxt],      A + arow + kk + ls);
      async_copy_b128(aoff[nxt] + 16, A + arow + kk + ls + 8);
      async_copy_b128(boff[nxt],      W + brow + kk + ls);
      async_copy_b128(boff[nxt] + 16, W + brow + kk + ls + 8);
    }

    v16bf bfr[2];
#pragma unroll
    for (int j = 0; j < 2; ++j)
      bfr[j] = load_b_frag(&Bs[cur][(wn * 32 + j * 16) * 32], 32);
#pragma unroll
    for (int i = 0; i < 4; ++i) {
      v16bf afr = load_a_frag(&As[cur][(wm * 64 + i * 16) * 32], 32);
#pragma unroll
      for (int j = 0; j < 2; ++j)
        acc[i][j] = wmma_bf16(afr, bfr[j], acc[i][j]);
    }
  }

  // epilogue: +bias, +residual, store fp32 or bf16
#pragma unroll
  for (int j = 0; j < 2; ++j) {
    const int gn = nTile + wn * 32 + j * 16 + nn;
    const float bv = bias ? bias[gn] : 0.f;
#pragma unroll
    for (int i = 0; i < 4; ++i) {
#pragma unroll
      for (int r = 0; r < 8; ++r) {
        const int gm = mTile + wm * 64 + i * 16 + r + 8 * half;
        const size_t idx = (size_t)gm * N + gn;
        float v = acc[i][j][r] + bv;
        if (res)  v += res[idx];
        if (outf) outf[idx] = v;
        else      outh[idx] = f2bf(v);
      }
    }
  }
}

// ---------------------------------------------------------------------------
// Flash attention: block = (batch b, head h, 64-row q-tile). d=64, N=1024.
// qkv: (B*1024) x 1536 bf16; head h channels: q=h*192, k=h*192+64, v=h*192+128
// Output: (B*1024) x 512 bf16, channel = h*64 + d.
// Q/K tiles staged with async-to-LDS; V transposed through VGPRs.
// ---------------------------------------------------------------------------
__global__ __launch_bounds__(256) void attn_kernel(
    const unsigned short* __restrict__ qkv,
    unsigned short* __restrict__ obuf) {
  __shared__ unsigned short Qs[64 * 64];   // [q][d]
  __shared__ unsigned short Ks[64 * 64];   // [key][d]
  __shared__ unsigned short Vt[64 * 64];   // [d][key]  (transposed)
  __shared__ unsigned short Ps[64 * 64];   // [q][key]
  __shared__ float m_run[64], l_run[64], alpha_s[64], mnew_s[64];
  __shared__ float part[64][2];

  const int t = threadIdx.x;
  const int w = t >> 5;
  const int lane = t & 31;
  const int half = lane >> 4;
  const int nn   = lane & 15;
  const int strip = w >> 1;      // 0..3: 16-row strip of the 64 q rows
  const int ch    = w & 1;       // 0..1: 32-wide column half

  const int qt = blockIdx.x;     // q tile 0..15
  const int hd = blockIdx.y;     // head 0..7
  const int bb = blockIdx.z;     // batch 0..7

  const size_t rowbase = (size_t)(bb * 1024 + qt * 64);
  const size_t kvbase  = (size_t)(bb * 1024);
  const int qc = hd * 192, kc = hd * 192 + 64, vc = hd * 192 + 128;

  const int srow = t >> 2, seg = (t & 3) * 16;

  // stage Q tile (async)
  {
    const unsigned short* gq = qkv + (rowbase + srow) * 1536 + qc + seg;
    unsigned int qoff = lds_off_of(&Qs[srow * 64 + seg]);
    async_copy_b128(qoff,      gq);
    async_copy_b128(qoff + 16, gq + 8);
  }
  if (t < 64) { m_run[t] = -3.0e38f; l_run[t] = 0.f; }

  const v8f vz = {0.f, 0.f, 0.f, 0.f, 0.f, 0.f, 0.f, 0.f};
  v8f Oacc[2]; Oacc[0] = vz; Oacc[1] = vz;

  for (int tile = 0; tile < 16; ++tile) {
    __syncthreads();                      // protect Ks/Vt/Ps/part reuse
    {                                     // stage K async; V via VGPR transpose
      const unsigned short* gk = qkv + (kvbase + tile * 64 + srow) * 1536 + kc + seg;
      unsigned int koff = lds_off_of(&Ks[srow * 64 + seg]);
      async_copy_b128(koff,      gk);
      async_copy_b128(koff + 16, gk + 8);
      const uint4* vp = (const uint4*)(qkv + (kvbase + tile * 64 + srow) * 1536 + vc + seg);
      uint4 vv0 = vp[0], vv1 = vp[1];
      unsigned short tmp[16];
      *(uint4*)&tmp[0] = vv0; *(uint4*)&tmp[8] = vv1;
#pragma unroll
      for (int e = 0; e < 16; ++e) Vt[(seg + e) * 64 + srow] = tmp[e];
    }
    wait_async0();                        // Q (iter 0) + K writes landed
    __syncthreads();

    // S = scale * Q K^T : this wave covers rows strip*16.., cols ch*32 + j*16
    v8f sf[2]; sf[0] = vz; sf[1] = vz;
#pragma unroll
    for (int kk = 0; kk < 64; kk += 32) {
      v16bf a = load_a_frag(&Qs[(strip * 16) * 64 + kk], 64);
#pragma unroll
      for (int j = 0; j < 2; ++j) {
        v16bf bfr = load_b_frag(&Ks[(ch * 32 + j * 16) * 64 + kk], 64);
        sf[j] = wmma_bf16(a, bfr, sf[j]);
      }
    }
#pragma unroll
    for (int j = 0; j < 2; ++j)
#pragma unroll
      for (int r = 0; r < 8; ++r) sf[j][r] *= 0.125f;   // d^-0.5

    // partial row-max across this wave's 32 columns
    float rm[8];
#pragma unroll
    for (int r = 0; r < 8; ++r) rm[r] = fmaxf(sf[0][r], sf[1][r]);
#pragma unroll
    for (int o = 8; o; o >>= 1)
#pragma unroll
      for (int r = 0; r < 8; ++r) rm[r] = fmaxf(rm[r], __shfl_xor(rm[r], o, 32));
    if (nn == 0) {
#pragma unroll
      for (int r = 0; r < 8; ++r) part[strip * 16 + r + 8 * half][ch] = rm[r];
    }
    __syncthreads();

    if (t < 64) {  // online-softmax stats
      const float mt = fmaxf(part[t][0], part[t][1]);
      const float mo = m_run[t];
      const float mn = fmaxf(mo, mt);
      mnew_s[t]  = mn;
      alpha_s[t] = __expf(mo - mn);
    }
    __syncthreads();

    // P = exp(S - m_new): store bf16 tile + partial row sums
    float rs[8];
#pragma unroll
    for (int r = 0; r < 8; ++r) rs[r] = 0.f;
#pragma unroll
    for (int j = 0; j < 2; ++j) {
#pragma unroll
      for (int r = 0; r < 8; ++r) {
        const int row = strip * 16 + r + 8 * half;
        const float p = __expf(sf[j][r] - mnew_s[row]);
        rs[r] += p;
        Ps[row * 64 + ch * 32 + j * 16 + nn] = f2bf(p);
      }
    }
#pragma unroll
    for (int o = 8; o; o >>= 1)
#pragma unroll
      for (int r = 0; r < 8; ++r) rs[r] += __shfl_xor(rs[r], o, 32);
    if (nn == 0) {
#pragma unroll
      for (int r = 0; r < 8; ++r) part[strip * 16 + r + 8 * half][ch] = rs[r];
    }
    __syncthreads();

    if (t < 64) {
      l_run[t] = l_run[t] * alpha_s[t] + part[t][0] + part[t][1];
      m_run[t] = mnew_s[t];
    }

    // O = O*alpha + P @ V  (this wave: rows strip*16.., d cols ch*32 + j*16)
#pragma unroll
    for (int j = 0; j < 2; ++j)
#pragma unroll
      for (int r = 0; r < 8; ++r)
        Oacc[j][r] *= alpha_s[strip * 16 + r + 8 * half];
#pragma unroll
    for (int kk = 0; kk < 64; kk += 32) {
      v16bf a = load_a_frag(&Ps[(strip * 16) * 64 + kk], 64);
#pragma unroll
      for (int j = 0; j < 2; ++j) {
        v16bf bfr = load_b_frag(&Vt[(ch * 32 + j * 16) * 64 + kk], 64);
        Oacc[j] = wmma_bf16(a, bfr, Oacc[j]);
      }
    }
  }
  __syncthreads();

  // normalize by l and write bf16, channel = hd*64 + ch*32 + j*16 + nn
#pragma unroll
  for (int j = 0; j < 2; ++j) {
#pragma unroll
    for (int r = 0; r < 8; ++r) {
      const int row = strip * 16 + r + 8 * half;
      const float ov = Oacc[j][r] / l_run[row];
      obuf[(rowbase + row) * 512 + hd * 64 + ch * 32 + j * 16 + nn] = f2bf(ov);
    }
  }
}

// ---------------------------------------------------------------------------
// Depthwise 3x3 conv (SAME, groups=C) + bias + exact GELU, bf16 in/out
// in/out: (B*1024) x 2048, spatial n = hh*32 + ww
// ---------------------------------------------------------------------------
__global__ __launch_bounds__(256) void dwconv_gelu_kernel(
    const unsigned short* __restrict__ in, const float* __restrict__ wgt,
    const float* __restrict__ bias, unsigned short* __restrict__ out) {
  const int c = blockIdx.x * 256 + threadIdx.x;   // 0..2047
  const int n = blockIdx.y;                       // 0..1023
  const int b = blockIdx.z;                       // 0..7
  const int hh = n >> 5, ww = n & 31;
  float acc = bias[c];
  const float* wc = wgt + c * 9;
#pragma unroll
  for (int i = 0; i < 3; ++i) {
    const int y = hh + i - 1;
    if (y < 0 || y > 31) continue;
#pragma unroll
    for (int j = 0; j < 3; ++j) {
      const int x2 = ww + j - 1;
      if (x2 < 0 || x2 > 31) continue;
      const int nsrc = (y << 5) + x2;
      acc += bf2f(in[(size_t)(b * 1024 + nsrc) * 2048 + c]) * wc[i * 3 + j];
    }
  }
  const float ge = 0.5f * acc * (1.f + erff(acc * 0.70710678118654752f));
  out[(size_t)(b * 1024 + n) * 2048 + c] = f2bf(ge);
}

// ---------------------------------------------------------------------------
// Host-side orchestration
// ---------------------------------------------------------------------------
extern "C" void kernel_launch(void* const* d_in, const int* in_sizes, int n_in,
                              void* d_out, int out_size, void* d_ws, size_t ws_size,
                              hipStream_t stream) {
  (void)in_sizes; (void)n_in; (void)out_size; (void)ws_size;
  const float* x      = (const float*)d_in[0];
  const float* ln1_g  = (const float*)d_in[1];
  const float* ln1_b  = (const float*)d_in[2];
  const float* qkv_w  = (const float*)d_in[3];
  const float* proj_w = (const float*)d_in[4];
  const float* proj_b = (const float*)d_in[5];
  const float* ln2_g  = (const float*)d_in[6];
  const float* ln2_b  = (const float*)d_in[7];
  const float* fc1_w  = (const float*)d_in[8];
  const float* fc1_b  = (const float*)d_in[9];
  const float* dw_w   = (const float*)d_in[10];
  const float* dw_b   = (const float*)d_in[11];
  const float* fc2_w  = (const float*)d_in[12];
  const float* fc2_b  = (const float*)d_in[13];
  float* out = (float*)d_out;

  // workspace layout (bytes); FC1 reuses the QKV+attnO region (dead by then)
  char* ws = (char*)d_ws;
  unsigned short* WQ  = (unsigned short*)(ws + 0);          // 1536x512 bf16
  unsigned short* WP  = (unsigned short*)(ws + 1572864);    //  512x512 bf16
  unsigned short* W1  = (unsigned short*)(ws + 2097152);    // 2048x512 bf16
  unsigned short* W2  = (unsigned short*)(ws + 4194304);    //  512x2048 bf16
  unsigned short* Hb  = (unsigned short*)(ws + 6291456);    // 8192x512 bf16 (ln out)
  unsigned short* QKV = (unsigned short*)(ws + 14680064);   // 8192x1536 bf16
  unsigned short* Ob  = (unsigned short*)(ws + 39845888);   // 8192x512 bf16
  unsigned short* FC1 = (unsigned short*)(ws + 14680064);   // 8192x2048 bf16 (reuse)
  float*          X1  = (float*)(ws + 48234496);            // 8192x512 fp32
  unsigned short* ACT = (unsigned short*)(ws + 65011712);   // 8192x2048 bf16

  // 0) weights -> bf16
  cvt_bf16_kernel<<<(786432  + 255) / 256, 256, 0, stream>>>(qkv_w,  WQ, 786432);
  cvt_bf16_kernel<<<(262144  + 255) / 256, 256, 0, stream>>>(proj_w, WP, 262144);
  cvt_bf16_kernel<<<(1048576 + 255) / 256, 256, 0, stream>>>(fc1_w,  W1, 1048576);
  cvt_bf16_kernel<<<(1048576 + 255) / 256, 256, 0, stream>>>(fc2_w,  W2, 1048576);

  // 1) LN1 -> bf16
  ln_bf16_kernel<<<8192, 256, 0, stream>>>(x, ln1_g, ln1_b, Hb);
  // 2) qkv = h @ qkv_w^T         (M=8192, N=1536, K=512) -> bf16
  gemm_bf16<<<dim3(12, 64), 256, 0, stream>>>(Hb, WQ, nullptr, nullptr,
                                              nullptr, QKV, 8192, 1536, 512);
  // 3) flash attention -> bf16 (B,N,C) with head-concat channels
  attn_kernel<<<dim3(16, 8, 8), 256, 0, stream>>>(QKV, Ob);
  // 4) x1 = x + o @ proj_w^T + proj_b   (fp32)
  gemm_bf16<<<dim3(4, 64), 256, 0, stream>>>(Ob, WP, proj_b, x,
                                             X1, nullptr, 8192, 512, 512);
  // 5) LN2 -> bf16
  ln_bf16_kernel<<<8192, 256, 0, stream>>>(X1, ln2_g, ln2_b, Hb);
  // 6) fc1 (M=8192, N=2048, K=512) -> bf16
  gemm_bf16<<<dim3(16, 64), 256, 0, stream>>>(Hb, W1, fc1_b, nullptr,
                                              nullptr, FC1, 8192, 2048, 512);
  // 7) dwconv3x3 + GELU -> bf16
  dwconv_gelu_kernel<<<dim3(8, 1024, 8), 256, 0, stream>>>(FC1, dw_w, dw_b, ACT);
  // 8) out = x1 + act @ fc2_w^T + fc2_b   (M=8192, N=512, K=2048, fp32)
  gemm_bf16<<<dim3(4, 64), 256, 0, stream>>>(ACT, W2, fc2_b, X1,
                                             out, nullptr, 8192, 512, 2048);
}
